// TopKMoE_21715354648868
// MI455X (gfx1250) — matrix operand order repaired and checked by reading, not compiled, not run on hardware
//
#include <hip/hip_runtime.h>
#include <hip/hip_bf16.h>

// ---------------------------------------------------------------------------
// Top-2 MoE for MI455X (gfx1250, wave32).
// Sparse dispatch + bf16x3 WMMA GEMMs + async-to-LDS double-buffered staging.
// ---------------------------------------------------------------------------

typedef __attribute__((ext_vector_type(16))) __bf16 v16bf;
typedef __attribute__((ext_vector_type(8)))  float  v8f;
typedef __attribute__((ext_vector_type(4)))  __bf16 bf16x4;

#define T_TOK 8192
#define DM    1024
#define DF    4096
#define NE    8

#define BM 128
#define BN 128
#define BK 32
#define LDA 40                 // 32 + 8 pad -> 80B row stride (16B aligned rows)
#define BUFBYTES (2 * BM * LDA * 2)   // one double-buffer stage of a tile pair (20480B)

// ---- bf16 helpers (bit-exact RNE) ----
static __device__ __forceinline__ __bf16 f32_to_bf16(float f) {
    unsigned u = __float_as_uint(f);
    unsigned r = (u + 0x7FFFu + ((u >> 16) & 1u)) >> 16;
    unsigned short s = (unsigned short)r;
    return __builtin_bit_cast(__bf16, s);
}
static __device__ __forceinline__ float bf16_to_f32(__bf16 b) {
    unsigned short s = __builtin_bit_cast(unsigned short, b);
    return __uint_as_float(((unsigned)s) << 16);
}
static __device__ __forceinline__ void split_bf16(float v, __bf16& hi, __bf16& lo) {
    hi = f32_to_bf16(v);
    lo = f32_to_bf16(v - bf16_to_f32(hi));
}

// ---- gfx1250 async copy: global -> LDS, 16B per lane, tracked by ASYNCcnt ----
static __device__ __forceinline__ void async_b128(unsigned lds_addr, const __bf16* g) {
    asm volatile("global_load_async_to_lds_b128 %0, %1, off"
                 :: "v"(lds_addr), "v"(g) : "memory");
}
static __device__ __forceinline__ void wait_async0() {
    asm volatile("s_wait_asynccnt 0x0" ::: "memory");
}

// ---- fragment loads from LDS, contiguous k order (k = 16*(lane/16) + s) for both A and B ----
static __device__ __forceinline__ v16bf load_frag_a(const __bf16* row, int lane) {
    int kb = (lane & 16) ? 16 : 0;
    bf16x4 c0 = *(const bf16x4*)(row + kb);
    bf16x4 c1 = *(const bf16x4*)(row + kb + 4);
    bf16x4 c2 = *(const bf16x4*)(row + kb + 8);
    bf16x4 c3 = *(const bf16x4*)(row + kb + 12);
    v16bf f;
#pragma unroll
    for (int i = 0; i < 4; ++i) { f[i] = c0[i]; f[4+i] = c1[i]; f[8+i] = c2[i]; f[12+i] = c3[i]; }
    return f;
}
static __device__ __forceinline__ v16bf load_frag_b(const __bf16* row, int lane) {
    int ks = (lane & 16) ? 16 : 0;
    v16bf f;
#pragma unroll
    for (int c = 0; c < 4; ++c) {
        bf16x4 v = *(const bf16x4*)(row + ks + c * 4);
#pragma unroll
        for (int i = 0; i < 4; ++i) f[c*4 + i] = v[i];
    }
    return f;
}

// ---------------------------------------------------------------------------
// init: zero output accumulator + per-expert counters
// ---------------------------------------------------------------------------
__global__ void moe_init(float* __restrict__ out, int* __restrict__ counts) {
    long i = (long)blockIdx.x * blockDim.x + threadIdx.x;
    const long n = (long)T_TOK * DM;
    for (long j = i; j < n; j += (long)gridDim.x * blockDim.x) out[j] = 0.f;
    if (i < NE) counts[i] = 0;
}

// ---------------------------------------------------------------------------
// preconvert x: f32 -> bf16 hi/lo planes (same layout)
// ---------------------------------------------------------------------------
__global__ void conv_x(const float* __restrict__ x, __bf16* __restrict__ xh,
                       __bf16* __restrict__ xl) {
    long i = (long)blockIdx.x * blockDim.x + threadIdx.x;
    const long n = (long)T_TOK * DM;
    for (; i < n; i += (long)gridDim.x * blockDim.x) {
        __bf16 hi, lo; split_bf16(x[i], hi, lo);
        xh[i] = hi; xl[i] = lo;
    }
}

// ---------------------------------------------------------------------------
// preconvert weights: [E][K][N] f32 -> [E][N][K] bf16 hi/lo planes (LDS transpose)
// ---------------------------------------------------------------------------
__global__ void conv_w(const float* __restrict__ src, __bf16* __restrict__ dh,
                       __bf16* __restrict__ dl, int K, int N) {
    __shared__ float tile[32][33];
    long ebase = (long)blockIdx.z * K * N;
    int k0 = blockIdx.x * 32;
    int n0 = blockIdx.y * 32;
    int tn = threadIdx.x & 31;
    int tk = threadIdx.x >> 5;                    // 0..7
#pragma unroll
    for (int i = 0; i < 32; i += 8)
        tile[tk + i][tn] = src[ebase + (long)(k0 + tk + i) * N + n0 + tn];
    __syncthreads();
    int wk = threadIdx.x & 31;
    int wn = threadIdx.x >> 5;
#pragma unroll
    for (int i = 0; i < 32; i += 8) {
        __bf16 hi, lo; split_bf16(tile[wk][wn + i], hi, lo);
        long di = ebase + (long)(n0 + wn + i) * K + k0 + wk;
        dh[di] = hi; dl[di] = lo;
    }
}

// ---------------------------------------------------------------------------
// router: one wave per token. logits -> softmax(f32) -> top-2 -> dispatch lists
// ---------------------------------------------------------------------------
__global__ void moe_router(const float* __restrict__ x, const float* __restrict__ rw,
                           float* __restrict__ probs, int* __restrict__ counts,
                           int* __restrict__ toks, float* __restrict__ wts) {
    int wave = threadIdx.x >> 5;
    int lane = threadIdx.x & 31;
    int t = blockIdx.x * 8 + wave;
    if (t >= T_TOK) return;
    const float* xr = x + (long)t * DM;
    float acc[NE];
#pragma unroll
    for (int e = 0; e < NE; ++e) acc[e] = 0.f;
    for (int d = lane; d < DM; d += 32) {
        float xv = xr[d];
        const float* wr = rw + d * NE;
#pragma unroll
        for (int e = 0; e < NE; ++e) acc[e] += xv * wr[e];
    }
#pragma unroll
    for (int off = 16; off > 0; off >>= 1)
#pragma unroll
        for (int e = 0; e < NE; ++e) acc[e] += __shfl_xor(acc[e], off, 32);
    if (lane == 0) {
        float mx = acc[0];
#pragma unroll
        for (int e = 1; e < NE; ++e) mx = fmaxf(mx, acc[e]);
        float p[NE], s = 0.f;
#pragma unroll
        for (int e = 0; e < NE; ++e) { p[e] = __expf(acc[e] - mx); s += p[e]; }
        float inv = 1.f / s;
        int i1 = 0; float v1 = -1.f;
#pragma unroll
        for (int e = 0; e < NE; ++e) {
            p[e] *= inv;
            probs[t * NE + e] = p[e];
            if (p[e] > v1) { v1 = p[e]; i1 = e; }     // strict > : first index wins ties
        }
        int i2 = 0; float v2 = -2.f;
#pragma unroll
        for (int e = 0; e < NE; ++e)
            if (e != i1 && p[e] > v2) { v2 = p[e]; i2 = e; }
        float wsum = v1 + v2 + 1e-9f;
        int pos1 = atomicAdd(&counts[i1], 1);
        toks[i1 * T_TOK + pos1] = t; wts[i1 * T_TOK + pos1] = v1 / wsum;
        int pos2 = atomicAdd(&counts[i2], 1);
        toks[i2 * T_TOK + pos2] = t; wts[i2 * T_TOK + pos2] = v2 / wsum;
    }
}

// ---------------------------------------------------------------------------
// aux loss: deterministic fixed-order reduction, single block
// ---------------------------------------------------------------------------
__global__ void moe_aux(const float* __restrict__ probs, float* __restrict__ out_aux) {
    __shared__ float sp[256 * NE];
    __shared__ float sc[256 * NE];
    float lp[NE], lc[NE];
#pragma unroll
    for (int e = 0; e < NE; ++e) { lp[e] = 0.f; lc[e] = 0.f; }
    for (int t = threadIdx.x; t < T_TOK; t += 256) {
        const float* pr = probs + t * NE;
        int am = 0; float bv = pr[0];
#pragma unroll
        for (int e = 0; e < NE; ++e) {
            float v = pr[e];
            lp[e] += v;
            if (v > bv) { bv = v; am = e; }
        }
        lc[am] += 1.f;
    }
#pragma unroll
    for (int e = 0; e < NE; ++e) { sp[threadIdx.x * NE + e] = lp[e]; sc[threadIdx.x * NE + e] = lc[e]; }
    __syncthreads();
    if (threadIdx.x == 0) {
        float aux = 0.f;
        for (int e = 0; e < NE; ++e) {
            float ps = 0.f, cs = 0.f;
            for (int i = 0; i < 256; ++i) { ps += sp[i * NE + e]; cs += sc[i * NE + e]; }
            aux += (cs / (float)T_TOK) * (ps / (float)T_TOK);
        }
        out_aux[0] = (float)NE * aux;
    }
}

// ---------------------------------------------------------------------------
// Unified expert GEMM (bf16x3 WMMA, async double-buffered LDS staging).
//   MODE 0: h = silu(x[toks] @ w1t)   A = x planes (token gather), K = DM
//   MODE 1: out[tok] += (h @ w2t)*wt  A = h planes (identity rows), K = DF
// B planes are pre-transposed [N][K] bf16, already offset to expert e.
// ---------------------------------------------------------------------------
template <int MODE>
__global__ void __launch_bounds__(256) moe_gemm(
    const __bf16* __restrict__ Ah, const __bf16* __restrict__ Al,
    const __bf16* __restrict__ Bh, const __bf16* __restrict__ Bl,
    const int* __restrict__ counts, const int* __restrict__ toks,
    const float* __restrict__ wts,
    __bf16* __restrict__ h_hi, __bf16* __restrict__ h_lo,
    float* __restrict__ out, int e) {
    constexpr int KD = (MODE == 0) ? DM : DF;
    int cnt = counts[e];
    int row0 = blockIdx.x * BM;
    if (row0 >= cnt) return;                      // uniform exit: EXEC all-1 at WMMA
    int col0 = blockIdx.y * BN;
    const int*   etoks = toks + e * T_TOK;
    const float* ewts  = wts + e * T_TOK;

    __shared__ __align__(16) __bf16 As[2][2][BM][LDA];   // [buf][plane][row][K]
    __shared__ __align__(16) __bf16 Bs[2][2][BN][LDA];

    int tid = threadIdx.x, lane = tid & 31, wv = tid >> 5;
    int wm = (wv & 1) * 64, wn = (wv >> 1) * 32;

    // ---- async staging setup: 8 x b128 chunks per thread per K-step ----
    int r0 = tid >> 2;                 // 0..63
    int ck = (tid & 3) * 8;            // chunk offset within row, elements (16B)
    int rows[2] = { r0, r0 + 64 };
    long arow[2];
#pragma unroll
    for (int j = 0; j < 2; ++j) {
        int p = row0 + rows[j];
        int pc = p < cnt ? p : cnt - 1;
        arow[j] = (MODE == 0) ? (long)etoks[pc] : (long)pc;
    }
    const __bf16* gA[4]; const __bf16* gB[4];
    unsigned lA[4], lB[4];
#pragma unroll
    for (int pl = 0; pl < 2; ++pl)
#pragma unroll
        for (int j = 0; j < 2; ++j) {
            const __bf16* ap = pl ? Al : Ah;
            const __bf16* bp = pl ? Bl : Bh;
            gA[pl * 2 + j] = ap + arow[j] * KD + ck;
            gB[pl * 2 + j] = bp + (long)(col0 + rows[j]) * KD + ck;
            lA[pl * 2 + j] = (unsigned)(unsigned long long)&As[0][pl][rows[j]][ck];
            lB[pl * 2 + j] = (unsigned)(unsigned long long)&Bs[0][pl][rows[j]][ck];
        }

    auto issue = [&](int buf, int kelem) {
        unsigned d = buf ? (unsigned)BUFBYTES : 0u;
#pragma unroll
        for (int i = 0; i < 4; ++i) async_b128(lA[i] + d, gA[i] + kelem);
#pragma unroll
        for (int i = 0; i < 4; ++i) async_b128(lB[i] + d, gB[i] + kelem);
    };

    v8f acc[4][2];
#pragma unroll
    for (int mi = 0; mi < 4; ++mi)
#pragma unroll
        for (int ni = 0; ni < 2; ++ni)
#pragma unroll
            for (int r = 0; r < 8; ++r) acc[mi][ni][r] = 0.f;

    issue(0, 0);
    const int nsteps = KD / BK;
    for (int s = 0; s < nsteps; ++s) {
        int cur = s & 1;
        wait_async0();                 // own async writes for buf `cur` complete
        __syncthreads();               // everyone's tile for `cur` complete
        if (s + 1 < nsteps) issue(cur ^ 1, (s + 1) * BK);   // prefetch overlaps compute

        v16bf a_hi[4], a_lo[4], b_hi[2], b_lo[2];
#pragma unroll
        for (int mi = 0; mi < 4; ++mi) {
            a_hi[mi] = load_frag_a(&As[cur][0][wm + mi * 16 + (lane & 15)][0], lane);
            a_lo[mi] = load_frag_a(&As[cur][1][wm + mi * 16 + (lane & 15)][0], lane);
        }
#pragma unroll
        for (int ni = 0; ni < 2; ++ni) {
            b_hi[ni] = load_frag_b(&Bs[cur][0][wn + ni * 16 + (lane & 15)][0], lane);
            b_lo[ni] = load_frag_b(&Bs[cur][1][wn + ni * 16 + (lane & 15)][0], lane);
        }
#pragma unroll
        for (int mi = 0; mi < 4; ++mi)
#pragma unroll
            for (int ni = 0; ni < 2; ++ni) {
                acc[mi][ni] = __builtin_amdgcn_wmma_f32_16x16x32_bf16(false, a_hi[mi], false, b_hi[ni], (short)0, acc[mi][ni], false, false);
                acc[mi][ni] = __builtin_amdgcn_wmma_f32_16x16x32_bf16(false, a_hi[mi], false, b_lo[ni], (short)0, acc[mi][ni], false, false);
                acc[mi][ni] = __builtin_amdgcn_wmma_f32_16x16x32_bf16(false, a_lo[mi], false, b_hi[ni], (short)0, acc[mi][ni], false, false);
            }
    }

    // ---- epilogue ----
    int lrow = (lane & 16) ? 8 : 0;
    int lcol = lane & 15;
#pragma unroll
    for (int mi = 0; mi < 4; ++mi)
#pragma unroll
        for (int r = 0; r < 8; ++r) {
            int p = row0 + wm + mi * 16 + lrow + r;
            if (p < cnt) {
                if (MODE == 0) {
#pragma unroll
                    for (int ni = 0; ni < 2; ++ni) {
                        int n = col0 + wn + ni * 16 + lcol;
                        float v = acc[mi][ni][r];
                        float sv = v / (1.f + __expf(-v));
                        __bf16 hi, lo; split_bf16(sv, hi, lo);
                        long idx = (long)p * DF + n;
                        h_hi[idx] = hi; h_lo[idx] = lo;
                    }
                } else {
                    int tok = etoks[p];
                    float wt = ewts[p];
#pragma unroll
                    for (int ni = 0; ni < 2; ++ni) {
                        int n = col0 + wn + ni * 16 + lcol;
                        long idx = (long)tok * DM + n;
                        out[idx] += acc[mi][ni][r] * wt;   // race-free: experts sequential
                    }
                }
            }
        }
}

// ---------------------------------------------------------------------------
// host-side orchestration
// ---------------------------------------------------------------------------
extern "C" void kernel_launch(void* const* d_in, const int* in_sizes, int n_in,
                              void* d_out, int out_size, void* d_ws, size_t ws_size,
                              hipStream_t stream) {
    const float* x  = (const float*)d_in[0];
    const float* rw = (const float*)d_in[1];
    const float* w1 = (const float*)d_in[2];
    const float* w2 = (const float*)d_in[3];
    float* out = (float*)d_out;

    // workspace carve-out (256B aligned slabs)
    char* ws = (char*)d_ws;
    size_t off = 0;
    auto carve = [&](size_t bytes) {
        char* p = ws + off;
        off = (off + bytes + 255) & ~(size_t)255;
        return p;
    };
    float*  probs  = (float*) carve((size_t)T_TOK * NE * 4);
    int*    counts = (int*)   carve(NE * 4);
    int*    toks   = (int*)   carve((size_t)NE * T_TOK * 4);
    float*  wts    = (float*) carve((size_t)NE * T_TOK * 4);
    __bf16* x_hi   = (__bf16*)carve((size_t)T_TOK * DM * 2);
    __bf16* x_lo   = (__bf16*)carve((size_t)T_TOK * DM * 2);
    __bf16* w1t_hi = (__bf16*)carve((size_t)NE * DM * DF * 2);
    __bf16* w1t_lo = (__bf16*)carve((size_t)NE * DM * DF * 2);
    __bf16* w2t_hi = (__bf16*)carve((size_t)NE * DF * DM * 2);
    __bf16* w2t_lo = (__bf16*)carve((size_t)NE * DF * DM * 2);
    __bf16* h_hi   = (__bf16*)carve((size_t)T_TOK * DF * 2);
    __bf16* h_lo   = (__bf16*)carve((size_t)T_TOK * DF * 2);

    moe_init<<<1024, 256, 0, stream>>>(out, counts);
    conv_x<<<4096, 256, 0, stream>>>(x, x_hi, x_lo);
    conv_w<<<dim3(DM / 32, DF / 32, NE), 256, 0, stream>>>(w1, w1t_hi, w1t_lo, DM, DF);
    conv_w<<<dim3(DF / 32, DM / 32, NE), 256, 0, stream>>>(w2, w2t_hi, w2t_lo, DF, DM);
    moe_router<<<T_TOK / 8, 256, 0, stream>>>(x, rw, probs, counts, toks, wts);
    moe_aux<<<1, 256, 0, stream>>>(probs, out + (long)T_TOK * DM);

    dim3 g1(T_TOK / BM, DF / BN);   // 64 x 32, most blocks exit early (sparse)
    dim3 g2(T_TOK / BM, DM / BN);   // 64 x 8
    for (int e = 0; e < NE; ++e) {
        const long w1o = (long)e * DM * DF;
        const long w2o = (long)e * DF * DM;
        moe_gemm<0><<<g1, 256, 0, stream>>>(x_hi, x_lo, w1t_hi + w1o, w1t_lo + w1o,
                                            counts, toks, wts, h_hi, h_lo, out, e);
        moe_gemm<1><<<g2, 256, 0, stream>>>(h_hi, h_lo, w2t_hi + w2o, w2t_lo + w2o,
                                            counts, toks, wts, h_hi, h_lo, out, e);
    }
}
